// ResonanceMapping_27161373180489
// MI455X (gfx1250) — matrix operand order repaired and checked by reading, not compiled, not run on hardware
//
#include <hip/hip_runtime.h>
#include <hip/hip_bf16.h>

// ---------------------------------------------------------------------------
// Problem constants (reference: B=2, S=192, D=256)
// ---------------------------------------------------------------------------
#define BB 2
#define SS 192
#define DD 256
#define RR (BB * SS) // 384 rows when batch is flattened

typedef __attribute__((ext_vector_type(16))) _Float16 v16h;
typedef __attribute__((ext_vector_type(8)))  _Float16 v8h;
typedef __attribute__((ext_vector_type(8)))  float    v8f;

// ---------------------------------------------------------------------------
// Device helpers
// ---------------------------------------------------------------------------
__device__ __forceinline__ float artanh_c(float z) {
    z = fminf(fmaxf(z, -1.0f + 1e-7f), 1.0f - 1e-7f);
    return 0.5f * __logf((1.0f + z) / (1.0f - z));
}

// Poincare-ball distance given g = x.y, x2 = |x|^2, y2 = |y|^2
__device__ __forceinline__ float hyp_dist(float g, float x2, float y2) {
    float a   = 1.0f - 2.0f * g + y2;
    float bb  = 1.0f - x2;
    float den = fmaxf(1.0f - 2.0f * g + x2 * y2, 1e-15f);
    float un2 = (a * a * x2 - 2.0f * a * bb * g + bb * bb * y2) / (den * den);
    float un  = sqrtf(fmaxf(un2, 0.0f));
    return 2.0f * artanh_c(un);
}

// Block reductions, blockDim.x == 256 everywhere they are used
__device__ __forceinline__ float block_sum(float v, float* sm) {
    int t = threadIdx.x;
    sm[t] = v; __syncthreads();
    for (int o = 128; o > 0; o >>= 1) {
        if (t < o) sm[t] += sm[t + o];
        __syncthreads();
    }
    float r = sm[0]; __syncthreads();
    return r;
}
__device__ __forceinline__ float block_min(float v, float* sm) {
    int t = threadIdx.x;
    sm[t] = v; __syncthreads();
    for (int o = 128; o > 0; o >>= 1) {
        if (t < o) sm[t] = fminf(sm[t], sm[t + o]);
        __syncthreads();
    }
    float r = sm[0]; __syncthreads();
    return r;
}

__device__ __forceinline__ float hashf(unsigned x) {
    x ^= x >> 16; x *= 0x7feb352dU; x ^= x >> 15; x *= 0x846ca68bU; x ^= x >> 16;
    return (float)x * (1.0f / 4294967296.0f) * 2.0f - 1.0f;
}

// ---------------------------------------------------------------------------
// WMMA fragment loaders (CDNA5 ISA 7.12.2, wave32, f16 operands)
// A fragment 16x32: lane-half h holds K = 16h + e (contiguous 16 halves)
// B fragment 32x16: lane-half h holds K = 16h + e (contiguous 16 halves)
// ---------------------------------------------------------------------------
__device__ __forceinline__ v16h load_afrag(const _Float16* rowp, int k0, int hl) {
    v8h lo = *(const v8h*)(rowp + k0 + hl * 16);
    v8h hi = *(const v8h*)(rowp + k0 + hl * 16 + 8);
    return __builtin_shufflevector(lo, hi, 0, 1, 2, 3, 4, 5, 6, 7,
                                           8, 9, 10, 11, 12, 13, 14, 15);
}
__device__ __forceinline__ v16h load_bfrag(const _Float16* rowp, int k0, int hl) {
    const v8h* p = (const v8h*)(rowp + k0 + hl * 16);
    v8h lo = p[0], hi = p[1];
    return __builtin_shufflevector(lo, hi, 0, 1, 2, 3, 4, 5, 6, 7,
                                           8, 9, 10, 11, 12, 13, 14, 15);
}

// ---------------------------------------------------------------------------
// WMMA GEMM (NT): C[M,N] = A[M,K] * B[N,K]^T (+ bias[N])
// f16 operands, f32 accumulate. One wave computes a 32x32 tile:
// 2 A-frags x 2 B-frags -> 4 v_wmma per K-step (operand reuse x2).
// grid = (N/32, M/32, batch), block = 32.
// ---------------------------------------------------------------------------
__global__ void gemm_nt_wmma(const _Float16* __restrict__ A,
                             const _Float16* __restrict__ Bm,
                             const float* __restrict__ bias,
                             float* __restrict__ C,
                             int M, int N, int K,
                             long sA, long sB, long sC) {
    int lane = threadIdx.x & 31;
    int hl   = lane >> 4;
    int l16  = lane & 15;
    int nt   = blockIdx.x * 32;
    int mt   = blockIdx.y * 32;
    const _Float16* Ab = A  + (long)blockIdx.z * sA;
    const _Float16* Bb = Bm + (long)blockIdx.z * sB;
    float*          Cb = C  + (long)blockIdx.z * sC;

    const _Float16* a0p = Ab + (long)(mt + l16) * K;
    const _Float16* a1p = Ab + (long)(mt + 16 + l16) * K;
    const _Float16* b0p = Bb + (long)(nt + l16) * K;
    const _Float16* b1p = Bb + (long)(nt + 16 + l16) * K;

    v8f acc00 = {0.f,0.f,0.f,0.f,0.f,0.f,0.f,0.f};
    v8f acc01 = acc00, acc10 = acc00, acc11 = acc00;

    for (int k0 = 0; k0 < K; k0 += 32) {
        __builtin_prefetch(a0p + k0 + 32, 0, 1);   // global_prefetch_b8
        v16h a0 = load_afrag(a0p, k0, hl);
        v16h a1 = load_afrag(a1p, k0, hl);
        v16h b0 = load_bfrag(b0p, k0, hl);
        v16h b1 = load_bfrag(b1p, k0, hl);
        acc00 = __builtin_amdgcn_wmma_f32_16x16x32_f16(false, a0, false, b0, (short)0, acc00, false, false);
        acc01 = __builtin_amdgcn_wmma_f32_16x16x32_f16(false, a0, false, b1, (short)0, acc01, false, false);
        acc10 = __builtin_amdgcn_wmma_f32_16x16x32_f16(false, a1, false, b0, (short)0, acc10, false, false);
        acc11 = __builtin_amdgcn_wmma_f32_16x16x32_f16(false, a1, false, b1, (short)0, acc11, false, false);
    }

    int c0 = nt + l16, c1 = nt + 16 + l16;
    float bv0 = bias ? bias[c0] : 0.0f;
    float bv1 = bias ? bias[c1] : 0.0f;
#pragma unroll
    for (int r = 0; r < 8; ++r) {
        int r0 = mt + r + (hl ? 8 : 0);
        int r1 = r0 + 16;
        Cb[(long)r0 * N + c0] = acc00[r] + bv0;
        Cb[(long)r0 * N + c1] = acc01[r] + bv1;
        Cb[(long)r1 * N + c0] = acc10[r] + bv0;
        Cb[(long)r1 * N + c1] = acc11[r] + bv1;
    }
}

// ---------------------------------------------------------------------------
// f32 -> f16 elementwise convert
// ---------------------------------------------------------------------------
__global__ void cvt_f16(const float* __restrict__ X, _Float16* __restrict__ Y, int n) {
    int i = blockIdx.x * 256 + threadIdx.x;
    if (i < n) Y[i] = (_Float16)X[i];
}

// ---------------------------------------------------------------------------
// Transpose + convert: X[b][Rx][Cx] f32 -> Y[b][Cx][Rx] f16.
// 16x16 LDS tile staged with CDNA5 async global->LDS copy (ASYNCcnt path).
// grid = (Cx/16, Rx/16, batch), block = 256.
// ---------------------------------------------------------------------------
__global__ void transpose_cvt16(const float* __restrict__ X, _Float16* __restrict__ Y,
                                int Rx, int Cx, long sX, long sY) {
    __shared__ float tile[16][17];
    int t  = threadIdx.x;
    int tx = t & 15, ty = t >> 4;
    int c0 = blockIdx.x * 16, r0 = blockIdx.y * 16;
    const float* Xb = X + (long)blockIdx.z * sX;
    _Float16*    Yb = Y + (long)blockIdx.z * sY;

    const float* src = Xb + (long)(r0 + ty) * Cx + c0 + tx;
    unsigned long long gaddr = (unsigned long long)src;
    unsigned ldsoff = (unsigned)(unsigned long long)(&tile[ty][tx]);
    // async DMA of one dword per lane into LDS (tracked by ASYNCcnt)
    asm volatile("global_load_async_to_lds_b32 %0, %1, off"
                 :: "v"(ldsoff), "v"(gaddr) : "memory");
    asm volatile("s_wait_asynccnt 0x0" ::: "memory");
    __syncthreads();

    Yb[(long)(c0 + ty) * Rx + r0 + tx] = (_Float16)tile[tx][ty];
}

// ---------------------------------------------------------------------------
// Row-wise kernels (blockDim = 256; one block per row)
// ---------------------------------------------------------------------------

// Xh = expmap0(clip_tangent(X)); Xh16 = f16 copy; Xsq = |Xh|^2
__global__ void hyp_rows(const float* __restrict__ X, float* __restrict__ Xh,
                         _Float16* __restrict__ Xh16, float* __restrict__ Xsq) {
    __shared__ float sm[256];
    int r = blockIdx.x, t = threadIdx.x;
    long idx = (long)r * DD + t;
    float x  = X[idx];
    float s1 = block_sum(x * x, sm);
    float n0 = sqrtf(s1) + 1e-8f;
    float sc = fminf(4.0f / n0, 1.0f);           // clip_tangent(max_norm=4)
    float xc = x * sc;
    float nc = fmaxf(sc * sqrtf(s1), 1e-15f);
    float th = tanhf(nc);
    float xh = th * xc / nc;
    Xh[idx]   = xh;
    Xh16[idx] = (_Float16)xh;
    if (t == 0) Xsq[r] = th * th;
}

// dist row -> w_unnorm, wn (f32 + f16), entropy/effective mass
__global__ void attn_row(const float* __restrict__ G, const float* __restrict__ q2,
                         const float* __restrict__ k2, float* __restrict__ w_un,
                         float* __restrict__ wn, _Float16* __restrict__ wn16,
                         float* __restrict__ effm) {
    __shared__ float sm[256];
    int b = blockIdx.y, q = blockIdx.x, t = threadIdx.x;
    bool act = t < SS;
    int  row = b * SS + q;
    long idx = (long)b * SS * SS + (long)q * SS + t;
    float x2 = q2[row];
    float d = 0.f;
    if (act) d = hyp_dist(G[idx], x2, k2[b * SS + t]);
    float dmin = block_min(act ? d : 1e30f, sm);
    float w    = act ? __expf(-(d - dmin)) : 0.f;
    float wsum = block_sum(w, sm);
    float wnv  = w / (wsum + 1e-8f);
    float ent  = block_sum(act ? -wnv * __logf(wnv + 1e-8f) : 0.f, sm);
    if (act) { w_un[idx] = w; wn[idx] = wnv; wn16[idx] = (_Float16)wnv; }
    if (t == 0) effm[row] = __expf(ent);
}

// Karcher flow per-pair scalars: R[q,s] = wn*c/den ; pA = sum_s R*alpha
__global__ void karcher_coef(const float* __restrict__ Gm, const float* __restrict__ mu2,
                             const float* __restrict__ k2, const float* __restrict__ wn,
                             _Float16* __restrict__ Rm16, float* __restrict__ pA) {
    __shared__ float sm[256];
    int b = blockIdx.y, q = blockIdx.x, t = threadIdx.x;
    bool act = t < SS;
    int  row = b * SS + q;
    long idx = (long)b * SS * SS + (long)q * SS + t;
    float x2 = mu2[row];
    float g = 0.f, y2 = 0.f, w = 0.f;
    if (act) { g = Gm[idx]; y2 = k2[b * SS + t]; w = wn[idx]; }
    float a   = 1.0f - 2.0f * g + y2;
    float bb  = 1.0f - x2;
    float den = fmaxf(1.0f - 2.0f * g + x2 * y2, 1e-15f);
    float un2 = (a * a * x2 - 2.0f * a * bb * g + bb * bb * y2) / (den * den);
    float un  = fmaxf(sqrtf(fmaxf(un2, 0.f)), 1e-15f);
    float l2  = fmaxf(1.0f - x2, 1e-15f);        // 2/lambda_x
    float c   = l2 * artanh_c(un) / un;
    float r   = act ? w * c / den : 0.f;
    if (act) Rm16[idx] = (_Float16)r;
    float p = block_sum(r * a, sm);
    if (t == 0) pA[row] = p;
}

// mu <- expmap(mu, clip(0.1 * (beta*T - pA*mu)))
__global__ void karcher_update(float* __restrict__ mu, _Float16* __restrict__ mu16,
                               float* __restrict__ mu2,
                               const float* __restrict__ T, const float* __restrict__ pA) {
    __shared__ float sm[256];
    int r = blockIdx.x, t = threadIdx.x;
    long idx = (long)r * DD + t;
    float x2 = mu2[r];
    float bb = 1.0f - x2;
    float m  = mu[idx];
    float vg = 0.1f * (bb * T[idx] - pA[r] * m);
    float s1 = block_sum(vg * vg, sm);
    float n0 = sqrtf(s1) + 1e-8f;
    float sc = fminf(4.0f / n0, 1.0f);
    float vc = vg * sc;
    float nv  = fmaxf(sc * sqrtf(s1), 1e-15f);
    float lam = 2.0f / fmaxf(1.0f - x2, 1e-15f);
    float th  = tanhf(lam * nv * 0.5f);
    float w   = th * vc / nv;                    // tanh(lam*n/2)*v/n
    float g   = block_sum(m * w, sm);            // mu . w
    float w2  = th * th;
    float num = (1.0f + 2.0f * g + w2) * m + (1.0f - x2) * w;
    float den = fmaxf(1.0f + 2.0f * g + x2 * w2, 1e-15f);
    float mn  = num / den;
    float s2  = block_sum(mn * mn, sm);
    mu[idx]   = mn;
    mu16[idx] = (_Float16)mn;
    if (t == 0) mu2[r] = s2;
}

__global__ void projx_k(const float* __restrict__ mu, float* __restrict__ cent,
                        _Float16* __restrict__ cent16, float* __restrict__ cent2) {
    __shared__ float sm[256];
    int r = blockIdx.x, t = threadIdx.x;
    long idx = (long)r * DD + t;
    float m  = mu[idx];
    float s1 = block_sum(m * m, sm);
    float n  = sqrtf(s1);
    float mx = 1.0f - 1e-5f;
    bool  ov = n > mx;
    float c  = ov ? m * mx / fmaxf(n, 1e-15f) : m;
    cent[idx]   = c;
    cent16[idx] = (_Float16)c;
    if (t == 0) cent2[r] = ov ? mx * mx : s1;
}

__global__ void tension_row(const float* __restrict__ Gc, const float* __restrict__ cent2,
                            const float* __restrict__ k2, const float* __restrict__ wn,
                            const float* __restrict__ effm, const float* __restrict__ tau,
                            float* __restrict__ var, float* __restrict__ ten) {
    __shared__ float sm[256];
    int b = blockIdx.y, q = blockIdx.x, t = threadIdx.x;
    bool act = t < SS;
    int  row = b * SS + q;
    long idx = (long)b * SS * SS + (long)q * SS + t;
    float x2 = cent2[row];
    float d = 0.f, w = 0.f;
    if (act) { d = hyp_dist(Gc[idx], x2, k2[b * SS + t]); w = wn[idx]; }
    float v = block_sum(w * d * d, sm);
    if (t == 0) { var[row] = v; ten[row] = v - tau[0] * effm[row]; }
}

__global__ void vrand_init(const float* __restrict__ cent, float* __restrict__ v,
                           _Float16* __restrict__ v16, float* __restrict__ cv) {
    __shared__ float sm[256];
    int r = blockIdx.x, t = threadIdx.x;
    long idx = (long)r * DD + t;
    float x  = hashf((unsigned)(idx * 2654435761u + 12345u));
    float s1 = block_sum(x * x, sm);
    float vn = x / fmaxf(sqrtf(s1), 1e-8f);
    v[idx]   = vn;
    v16[idx] = (_Float16)vn;
    float c = block_sum(cent[idx] * vn, sm);
    if (t == 0) cv[r] = c;
}

// power iteration per-pair scalars: proj = wk.v; E = B*proj; sA = sum A*proj
__global__ void power_coef(const float* __restrict__ Gc, const float* __restrict__ KV,
                           const float* __restrict__ cent2, const float* __restrict__ k2,
                           const float* __restrict__ w_un, const float* __restrict__ cv,
                           _Float16* __restrict__ E16, float* __restrict__ sA) {
    __shared__ float sm[256];
    int b = blockIdx.y, q = blockIdx.x, t = threadIdx.x;
    bool act = t < SS;
    int  row = b * SS + q;
    long idx = (long)b * SS * SS + (long)q * SS + t;
    float x2 = cent2[row];
    float g = 0.f, y2 = 0.f, w = 0.f, kv = 0.f;
    if (act) { g = Gc[idx]; y2 = k2[b * SS + t]; w = w_un[idx]; kv = KV[idx]; }
    float a   = 1.0f - 2.0f * g + y2;
    float bb  = 1.0f - x2;
    float den = fmaxf(1.0f - 2.0f * g + x2 * y2, 1e-15f);
    float un2 = (a * a * x2 - 2.0f * a * bb * g + bb * bb * y2) / (den * den);
    float un  = fmaxf(sqrtf(fmaxf(un2, 0.f)), 1e-15f);
    float l2  = fmaxf(1.0f - x2, 1e-15f);
    float c   = l2 * artanh_c(un) / un;
    float m   = act ? sqrtf(w + 1e-8f) * c / den : 0.f;
    float Aq  = m * a, Bq = m * bb;
    float proj = -Aq * cv[row] + Bq * kv;
    if (act) E16[idx] = (_Float16)(Bq * proj);
    float s = block_sum(Aq * proj, sm);
    if (t == 0) sA[row] = s;
}

__global__ void power_norm(const float* __restrict__ U, const float* __restrict__ sA,
                           const float* __restrict__ cent, float* __restrict__ v,
                           _Float16* __restrict__ v16, float* __restrict__ cv) {
    __shared__ float sm[256];
    int r = blockIdx.x, t = threadIdx.x;
    long idx = (long)r * DD + t;
    float x  = U[idx] - sA[r] * cent[idx];
    float s1 = block_sum(x * x, sm);
    float vn = x / fmaxf(sqrtf(s1), 1e-8f);
    v[idx]   = vn;
    v16[idx] = (_Float16)vn;
    float c = block_sum(cent[idx] * vn, sm);
    if (t == 0) cv[r] = c;
}

// fallback direction, h_comp, RK4 pitchfork ODE, gate fusion -> hf16
__global__ void final_row(const float* __restrict__ h, const float* __restrict__ v,
                          const float* __restrict__ cent, const float* __restrict__ cent2,
                          const float* __restrict__ qh, const float* __restrict__ q2,
                          const float* __restrict__ var, const float* __restrict__ ten,
                          const float* __restrict__ tau, const float* __restrict__ tsc,
                          _Float16* __restrict__ hf16) {
    __shared__ float sm[256];
    int r = blockIdx.x, t = threadIdx.x;
    long idx = (long)r * DD + t;
    float c2 = cent2[r];
    float wg = v[idx] / fmaxf(1.0f - c2, 1e-15f);      // transp0back
    float nq = fmaxf(sqrtf(q2[r]), 1e-15f);
    float nc = fmaxf(sqrtf(c2), 1e-15f);
    float fb = artanh_c(nq) * qh[idx] / nq - artanh_c(nc) * cent[idx] / nc;
    float fn = block_sum(fb * fb, sm);
    fb = fb / fmaxf(sqrtf(fn), 1e-8f);
    float wp = (var[r] > 1e-5f) ? wg : fb;

    float hv  = h[idx];
    float hn2 = block_sum(hv * hv, sm);
    float hn  = sqrtf(hn2);
    float hc  = asinhf(hn) * hv / (hn + 1e-8f);
    float x   = block_sum(hc * wp, sm);

    float T  = ten[r];
    float xi = (x == 0.0f) ? 1e-4f : x;
    const float dt = 0.5f;
#pragma unroll
    for (int i = 0; i < 4; ++i) {
        float k1 = dt * (T * xi - xi * xi * xi);
        float z  = xi + 0.5f * k1;
        float k2 = dt * (T * z - z * z * z);
        z = xi + 0.5f * k2;
        float k3 = dt * (T * z - z * z * z);
        z = xi + k3;
        float k4 = dt * (T * z - z * z * z);
        xi += (k1 + 2.f * k2 + 2.f * k3 + k4) * (1.0f / 6.0f);
    }

    float hp  = hc + (xi - x) * wp;
    float hts = hp * tsc[0];
    float s1  = block_sum(hts * hts, sm);
    float n0  = sqrtf(s1) + 1e-8f;
    float sc  = fminf(4.0f / n0, 1.0f);
    float ct  = hts * sc;
    float nct = fmaxf(sc * sqrtf(s1), 1e-15f);
    float th  = tanhf(nct);
    float hb  = artanh_c(fmaxf(th, 1e-15f)) * ct / nct;  // logmap0(expmap0(ct))
    float gate = fmaxf(tanhf(T / fmaxf(tau[0], 1e-3f)), 0.0f);
    hf16[idx] = (_Float16)((1.0f - gate) * hv + gate * hb);
}

// ---------------------------------------------------------------------------
// Host launch
// ---------------------------------------------------------------------------
extern "C" void kernel_launch(void* const* d_in, const int* in_sizes, int n_in,
                              void* d_out, int out_size, void* d_ws, size_t ws_size,
                              hipStream_t stream) {
    const float* q_in = (const float*)d_in[0];
    const float* k_in = (const float*)d_in[1];
    const float* v_in = (const float*)d_in[2];
    const float* Wq   = (const float*)d_in[3];
    const float* bq   = (const float*)d_in[4];
    const float* Wk   = (const float*)d_in[5];
    const float* bk   = (const float*)d_in[6];
    const float* Wv   = (const float*)d_in[7];
    const float* bv   = (const float*)d_in[8];
    const float* Wo   = (const float*)d_in[9];
    const float* bo   = (const float*)d_in[10];
    const float* tau  = (const float*)d_in[11];
    const float* tsc  = (const float*)d_in[12];
    float* out = (float*)d_out;

    // 64B-aligned bump allocator over workspace
    char*  base = (char*)d_ws;
    size_t off  = 0;
    auto alloc = [&](size_t bytes) {
        size_t a = (off + 63) & ~(size_t)63;
        void* p = base + a;
        off = a + bytes;
        return p;
    };
    const size_t RD  = (size_t)RR * DD;       // 98304
    const size_t SSB = (size_t)BB * SS * SS;  // 73728
    const size_t WW  = (size_t)DD * DD;       // 65536

    // f32 buffers
    float* q    = (float*)alloc(RD * 4);
    float* k    = (float*)alloc(RD * 4);
    float* vl   = (float*)alloc(RD * 4);
    float* qh   = (float*)alloc(RD * 4);
    float* kh   = (float*)alloc(RD * 4);
    float* mu   = (float*)alloc(RD * 4);
    float* tmat = (float*)alloc(RD * 4);   // T / U / mu0 temp
    float* cent = (float*)alloc(RD * 4);
    float* vv   = (float*)alloc(RD * 4);
    float* h    = (float*)alloc(RD * 4);
    float* gram = (float*)alloc(SSB * 4);  // G1 / Gm / KV
    float* Gc   = (float*)alloc(SSB * 4);
    float* w_un = (float*)alloc(SSB * 4);
    float* wn   = (float*)alloc(SSB * 4);
    float* q2   = (float*)alloc(RR * 4);
    float* k2   = (float*)alloc(RR * 4);
    float* effm = (float*)alloc(RR * 4);
    float* var  = (float*)alloc(RR * 4);
    float* ten  = (float*)alloc(RR * 4);
    float* rs   = (float*)alloc(RR * 4);   // pA / sA
    float* cv   = (float*)alloc(RR * 4);
    float* mu2  = (float*)alloc(RR * 4);
    float* c2   = (float*)alloc(RR * 4);
    // f16 operand buffers (16B-aligned fragment loads)
    _Float16* qin16  = (_Float16*)alloc(RD * 2);
    _Float16* kin16  = (_Float16*)alloc(RD * 2);
    _Float16* vin16  = (_Float16*)alloc(RD * 2);
    _Float16* Wq16   = (_Float16*)alloc(WW * 2);
    _Float16* Wk16   = (_Float16*)alloc(WW * 2);
    _Float16* Wv16   = (_Float16*)alloc(WW * 2);
    _Float16* Wo16   = (_Float16*)alloc(WW * 2);
    _Float16* qh16   = (_Float16*)alloc(RD * 2);
    _Float16* kh16   = (_Float16*)alloc(RD * 2);
    _Float16* khT16  = (_Float16*)alloc(RD * 2);  // [B,D,S]
    _Float16* kT16   = (_Float16*)alloc(RD * 2);  // [B,D,S]
    _Float16* vlT16  = (_Float16*)alloc(RD * 2);  // [B,D,S]
    _Float16* mu16   = (_Float16*)alloc(RD * 2);
    _Float16* cent16 = (_Float16*)alloc(RD * 2);
    _Float16* v16    = (_Float16*)alloc(RD * 2);
    _Float16* hf16   = (_Float16*)alloc(RD * 2);
    _Float16* wn16   = (_Float16*)alloc(SSB * 2);
    _Float16* coef16 = (_Float16*)alloc(SSB * 2); // R / E
    if (off > ws_size) return;  // insufficient scratch

    const long SD = (long)SS * DD, S2 = (long)SS * SS;
    dim3 b32(32), b256(256);
    dim3 gLin(DD / 32, RR / 32, 1);      // 8 x 12 : linear layers
    dim3 gGram(SS / 32, SS / 32, BB);    // 6 x 6 x 2 : Gram matrices
    dim3 gWsum(DD / 32, SS / 32, BB);    // 8 x 6 x 2 : weighted sums over S
    dim3 gRowS(SS, BB);
    dim3 gTr(DD / 16, SS / 16, BB);      // 16 x 12 x 2 : [S,D] -> [D,S] transpose
    dim3 gCvtRD((int)((RD + 255) / 256));
    dim3 gCvtW((int)((WW + 255) / 256));

    // f16 operand prep
    cvt_f16<<<gCvtRD, b256, 0, stream>>>(q_in, qin16, (int)RD);
    cvt_f16<<<gCvtRD, b256, 0, stream>>>(k_in, kin16, (int)RD);
    cvt_f16<<<gCvtRD, b256, 0, stream>>>(v_in, vin16, (int)RD);
    cvt_f16<<<gCvtW,  b256, 0, stream>>>(Wq, Wq16, (int)WW);
    cvt_f16<<<gCvtW,  b256, 0, stream>>>(Wk, Wk16, (int)WW);
    cvt_f16<<<gCvtW,  b256, 0, stream>>>(Wv, Wv16, (int)WW);
    cvt_f16<<<gCvtW,  b256, 0, stream>>>(Wo, Wo16, (int)WW);

    // Projections: q = q_in @ Wq^T + bq, etc. (flatten B*S rows)
    gemm_nt_wmma<<<gLin, b32, 0, stream>>>(qin16, Wq16, bq, q,  RR, DD, DD, 0, 0, 0);
    gemm_nt_wmma<<<gLin, b32, 0, stream>>>(kin16, Wk16, bk, k,  RR, DD, DD, 0, 0, 0);
    gemm_nt_wmma<<<gLin, b32, 0, stream>>>(vin16, Wv16, bv, vl, RR, DD, DD, 0, 0, 0);

    hyp_rows<<<RR, b256, 0, stream>>>(q, qh, qh16, q2);
    hyp_rows<<<RR, b256, 0, stream>>>(k, kh, kh16, k2);

    // transposed f16 operands for the weighted-sum GEMMs (all become NT)
    transpose_cvt16<<<gTr, b256, 0, stream>>>(k,  kT16,  SS, DD, SD, SD);
    transpose_cvt16<<<gTr, b256, 0, stream>>>(vl, vlT16, SS, DD, SD, SD);
    transpose_cvt16<<<gTr, b256, 0, stream>>>(kh, khT16, SS, DD, SD, SD);

    // Gram(q_hyp, k_hyp) -> distance attention weights
    gemm_nt_wmma<<<gGram, b32, 0, stream>>>(qh16, kh16, nullptr, gram, SS, SS, DD, SD, SD, S2);
    attn_row<<<gRowS, b256, 0, stream>>>(gram, q2, k2, w_un, wn, wn16, effm);

    // h = wn @ v ; mu0 = wn @ k -> mu_hyp
    gemm_nt_wmma<<<gWsum, b32, 0, stream>>>(wn16, vlT16, nullptr, h,    SS, DD, SS, S2, SD, SD);
    gemm_nt_wmma<<<gWsum, b32, 0, stream>>>(wn16, kT16,  nullptr, tmat, SS, DD, SS, S2, SD, SD);
    hyp_rows<<<RR, b256, 0, stream>>>(tmat, mu, mu16, mu2);

    // Karcher flow: 3 Riemannian SGD steps
    for (int it = 0; it < 3; ++it) {
        gemm_nt_wmma<<<gGram, b32, 0, stream>>>(mu16, kh16, nullptr, gram, SS, SS, DD, SD, SD, S2);
        karcher_coef<<<gRowS, b256, 0, stream>>>(gram, mu2, k2, wn, coef16, rs);
        gemm_nt_wmma<<<gWsum, b32, 0, stream>>>(coef16, khT16, nullptr, tmat, SS, DD, SS, S2, SD, SD);
        karcher_update<<<RR, b256, 0, stream>>>(mu, mu16, mu2, tmat, rs);
    }
    projx_k<<<RR, b256, 0, stream>>>(mu, cent, cent16, c2);

    // centroid-to-key distances -> variance / tension
    gemm_nt_wmma<<<gGram, b32, 0, stream>>>(cent16, kh16, nullptr, Gc, SS, SS, DD, SD, SD, S2);
    tension_row<<<gRowS, b256, 0, stream>>>(Gc, c2, k2, wn, effm, tau, var, ten);

    // power iteration for principal tangent direction
    vrand_init<<<RR, b256, 0, stream>>>(cent, vv, v16, cv);
    for (int it = 0; it < 3; ++it) {
        gemm_nt_wmma<<<gGram, b32, 0, stream>>>(v16, kh16, nullptr, gram, SS, SS, DD, SD, SD, S2);
        power_coef<<<gRowS, b256, 0, stream>>>(Gc, gram, c2, k2, w_un, cv, coef16, rs);
        gemm_nt_wmma<<<gWsum, b32, 0, stream>>>(coef16, khT16, nullptr, tmat, SS, DD, SS, S2, SD, SD);
        power_norm<<<RR, b256, 0, stream>>>(tmat, rs, cent, vv, v16, cv);
    }

    // bifurcation ODE + fusion, then output projection
    final_row<<<RR, b256, 0, stream>>>(h, vv, cent, c2, qh, q2, var, ten, tau, tsc, hf16);
    gemm_nt_wmma<<<gLin, b32, 0, stream>>>(hf16, Wo16, bo, out, RR, DD, DD, 0, 0, 0);
}
